// MixtureOfExperts_68925635166513
// MI455X (gfx1250) — matrix operand order repaired and checked by reading, not compiled, not run on hardware
//
#include <hip/hip_runtime.h>
#include <math.h>

typedef __attribute__((ext_vector_type(16))) _Float16 v16h;
typedef __attribute__((ext_vector_type(8)))  float    v8f;

#define T_TOK 4096
#define D_DIM 1024
#define H_DIM 4096
#define E_NUM 8
#define TM 128      // tokens per macro tile
#define TN 128      // output cols per macro tile
#define KB 32       // K chunk (WMMA f16 K)
#define LDA 40      // padded halfs per A row in LDS (80B -> conflict-free, 16B aligned)
#define LDB 40      // padded halfs per B row in LDS

// ---- WMMA fragment loaders (layouts per CDNA5 ISA 7.12.2, 16-bit A 16x32 / B 32x16) ----
__device__ __forceinline__ v16h load_a_frag(const _Float16* a_sh, int lane, int mbase) {
  int m  = mbase + (lane & 15);
  int k0 = (lane < 16) ? 0 : 8;    // f[0..7]  -> K = k0 + j
  int k1 = (lane < 16) ? 16 : 24;  // f[8..15] -> K = k1 + (j-8)
  v16h f;
  ((uint4*)&f)[0] = *(const uint4*)(a_sh + m * LDA + k0);
  ((uint4*)&f)[1] = *(const uint4*)(a_sh + m * LDA + k1);
  return f;
}

__device__ __forceinline__ v16h load_b_frag(const _Float16* b_sh, int lane, int nbase) {
  int n = nbase + (lane & 15);
  int k = (lane < 16) ? 0 : 8;     // f[0..7] -> K = k + j, f[8..15] -> K = k + 16 + (j-8)
  v16h f;
  ((uint4*)&f)[0] = *(const uint4*)(b_sh + n * LDB + k);
  ((uint4*)&f)[1] = *(const uint4*)(b_sh + n * LDB + k + 16);
  return f;
}

__device__ __forceinline__ void cvt_store8(_Float16* d, float4 f0, float4 f1) {
  d[0] = (_Float16)f0.x; d[1] = (_Float16)f0.y;
  d[2] = (_Float16)f0.z; d[3] = (_Float16)f0.w;
  d[4] = (_Float16)f1.x; d[5] = (_Float16)f1.y;
  d[6] = (_Float16)f1.z; d[7] = (_Float16)f1.w;
}

// ---- tiny utility kernels ----
__global__ void moe_zero_counts(int* counts) {
  if (threadIdx.x < E_NUM) counts[threadIdx.x] = 0;
}

__global__ __launch_bounds__(256) void moe_cvt_x(const float* __restrict__ x,
                                                 _Float16* __restrict__ xh) {
  size_t i = ((size_t)blockIdx.x * 256 + threadIdx.x) * 4;
  float4 v = *(const float4*)(x + i);
  xh[i + 0] = (_Float16)v.x;
  xh[i + 1] = (_Float16)v.y;
  xh[i + 2] = (_Float16)v.z;
  xh[i + 3] = (_Float16)v.w;
}

// ---- router: logits -> top-2 -> renormalized weights + per-expert compaction ----
__global__ __launch_bounds__(256) void moe_router(const float* __restrict__ x,
                                                  const float* __restrict__ rw,
                                                  int* __restrict__ counts,
                                                  int* __restrict__ pair_tok,
                                                  int* __restrict__ tok_slot,
                                                  float* __restrict__ tok_w) {
  const int lane = threadIdx.x & 31;
  const int t = blockIdx.x * 8 + (threadIdx.x >> 5);  // one wave32 per token
  const float* xr = x + (size_t)t * D_DIM;
  float acc[E_NUM];
#pragma unroll
  for (int e = 0; e < E_NUM; ++e) acc[e] = 0.f;
  for (int i = lane; i < D_DIM; i += 32) {
    float xv = xr[i];
#pragma unroll
    for (int e = 0; e < E_NUM; ++e) acc[e] += xv * rw[e * D_DIM + i];
  }
#pragma unroll
  for (int e = 0; e < E_NUM; ++e) {
    acc[e] += __shfl_down(acc[e], 16, 32);
    acc[e] += __shfl_down(acc[e], 8, 32);
    acc[e] += __shfl_down(acc[e], 4, 32);
    acc[e] += __shfl_down(acc[e], 2, 32);
    acc[e] += __shfl_down(acc[e], 1, 32);
  }
  if (lane == 0) {
    int e0 = 0;
#pragma unroll
    for (int e = 1; e < E_NUM; ++e)
      if (acc[e] > acc[e0]) e0 = e;
    int e1 = (e0 == 0) ? 1 : 0;
#pragma unroll
    for (int e = 0; e < E_NUM; ++e)
      if (e != e0 && acc[e] > acc[e1]) e1 = e;
    // renormalized top-2 softmax == 2-way logistic over the top-2 logits
    float w0 = 1.f / (1.f + expf(acc[e1] - acc[e0]));
    float w1 = 1.f - w0;
    int p0 = atomicAdd(&counts[e0], 1);
    int p1 = atomicAdd(&counts[e1], 1);
    pair_tok[e0 * T_TOK + p0] = t;
    pair_tok[e1 * T_TOK + p1] = t;
    tok_slot[2 * t + 0] = e0 * T_TOK + p0;
    tok_slot[2 * t + 1] = e1 * T_TOK + p1;
    tok_w[2 * t + 0] = w0;
    tok_w[2 * t + 1] = w1;
  }
}

__global__ void moe_scan(const int* __restrict__ counts, int* __restrict__ offsets) {
  int s = 0;
  for (int e = 0; e < E_NUM; ++e) { offsets[e] = s; s += counts[e]; }
  offsets[E_NUM] = s;
}

// ---- GEMM1: h = gelu(Xg @ W1[e]^T + b1[e]), gathered rows, f16 out ----
__global__ __launch_bounds__(256) void moe_gemm1(const _Float16* __restrict__ xh,
                                                 const float* __restrict__ w1,
                                                 const float* __restrict__ bias1,
                                                 const int* __restrict__ pair_tok,
                                                 const int* __restrict__ counts,
                                                 const int* __restrict__ offsets,
                                                 _Float16* __restrict__ hbuf) {
  __shared__ _Float16 a_sh[2][TM * LDA];   // double buffered: 2 x 10.0 KB
  __shared__ _Float16 b_sh[2][TN * LDB];   // double buffered: 2 x 10.0 KB
  const int e = blockIdx.x >> 5;       // T_TOK/TM == 32 row tiles per expert
  const int rt = blockIdx.x & 31;
  const int n_e = counts[e];
  const int rbase = rt * TM;
  if (rbase >= n_e) return;            // block-uniform early exit
  const int off_e = offsets[e];
  const int nbase = blockIdx.y * TN;
  const int tid = threadIdx.x;
  const int lane = tid & 31;
  const int wave = tid >> 5;
  const int wm = wave & 3;             // 4 waves along M (32 rows each)
  const int wn = wave >> 2;            // 2 waves along N (64 cols each)

  // A staging: thread -> (row, 16-half chunk); gather index clamped (rows >= n_e
  // produce garbage that is masked in the epilogue) -> no divergent branch.
  const int ar = tid >> 1;
  const int ac = (tid & 1) << 4;
  const int arow = (rbase + ar < n_e) ? (rbase + ar) : (n_e - 1);
  const int tok = pair_tok[e * T_TOK + arow];
  const _Float16* asrc = xh + (size_t)tok * D_DIM;
  // B staging: thread -> (n row, 16-wide k chunk); on-the-fly f32->f16
  const int bn = tid >> 1;             // 0..127
  const int bk = (tid & 1) << 4;       // 0 or 16
  const float* bsrc = w1 + ((size_t)e * H_DIM + nbase + bn) * D_DIM;

  const v8f vzero = {0.f, 0.f, 0.f, 0.f, 0.f, 0.f, 0.f, 0.f};
  v8f c[2][4];
#pragma unroll
  for (int mi = 0; mi < 2; ++mi)
#pragma unroll
    for (int ni = 0; ni < 4; ++ni) c[mi][ni] = vzero;

  // prologue: load + stage chunk 0 into buffer 0
  uint4 pa0 = *(const uint4*)(asrc + ac);
  uint4 pa1 = *(const uint4*)(asrc + ac + 8);
  float4 pb0 = *(const float4*)(bsrc + bk + 0);
  float4 pb1 = *(const float4*)(bsrc + bk + 4);
  float4 pb2 = *(const float4*)(bsrc + bk + 8);
  float4 pb3 = *(const float4*)(bsrc + bk + 12);
  *(uint4*)(a_sh[0] + ar * LDA + ac)     = pa0;
  *(uint4*)(a_sh[0] + ar * LDA + ac + 8) = pa1;
  cvt_store8(b_sh[0] + bn * LDB + bk,     pb0, pb1);
  cvt_store8(b_sh[0] + bn * LDB + bk + 8, pb2, pb3);
  __syncthreads();

  for (int kb = 0; kb < D_DIM; kb += KB) {
    const int p = (kb >> 5) & 1;
    const bool more = (kb + KB) < D_DIM;
    if (more) {  // issue next-chunk global loads early; latency hidden by WMMAs
      pa0 = *(const uint4*)(asrc + kb + KB + ac);
      pa1 = *(const uint4*)(asrc + kb + KB + ac + 8);
      pb0 = *(const float4*)(bsrc + kb + KB + bk + 0);
      pb1 = *(const float4*)(bsrc + kb + KB + bk + 4);
      pb2 = *(const float4*)(bsrc + kb + KB + bk + 8);
      pb3 = *(const float4*)(bsrc + kb + KB + bk + 12);
      __builtin_prefetch(bsrc + kb + 2 * KB, 0, 0);  // global_prefetch_b8
    }
    v16h a0 = load_a_frag(a_sh[p], lane, 32 * wm);
    v16h a1 = load_a_frag(a_sh[p], lane, 32 * wm + 16);
#pragma unroll
    for (int ni = 0; ni < 4; ++ni) {
      v16h bf = load_b_frag(b_sh[p], lane, 64 * wn + 16 * ni);
      c[0][ni] = __builtin_amdgcn_wmma_f32_16x16x32_f16(false, a0, false, bf, (short)0, c[0][ni], false, false);
      c[1][ni] = __builtin_amdgcn_wmma_f32_16x16x32_f16(false, a1, false, bf, (short)0, c[1][ni], false, false);
    }
    if (more) {  // stage into idle buffer; single barrier per chunk
      *(uint4*)(a_sh[p ^ 1] + ar * LDA + ac)     = pa0;
      *(uint4*)(a_sh[p ^ 1] + ar * LDA + ac + 8) = pa1;
      cvt_store8(b_sh[p ^ 1] + bn * LDB + bk,     pb0, pb1);
      cvt_store8(b_sh[p ^ 1] + bn * LDB + bk + 8, pb2, pb3);
      __syncthreads();
    }
  }

  // epilogue: C layout = lane N=lane&15, VGPR g -> M = g + (lane<16?0:8)
  const int nl = lane & 15;
  const int gh = (lane < 16) ? 0 : 8;
#pragma unroll
  for (int mi = 0; mi < 2; ++mi)
#pragma unroll
    for (int ni = 0; ni < 4; ++ni) {
      int col = nbase + 64 * wn + 16 * ni + nl;
      float bias = bias1[e * H_DIM + col];
#pragma unroll
      for (int g = 0; g < 8; ++g) {
        int r = rbase + 32 * wm + 16 * mi + gh + g;
        if (r < n_e) {
          float v = c[mi][ni][g] + bias;
          float gl = 0.5f * v * (1.0f + erff(v * 0.70710678118654752f));
          hbuf[(size_t)(off_e + r) * H_DIM + col] = (_Float16)gl;
        }
      }
    }
}

// ---- GEMM2: o = h @ W2[e]^T + b2[e] -> pairout rows (unweighted) ----
__global__ __launch_bounds__(256) void moe_gemm2(const _Float16* __restrict__ hbuf,
                                                 const float* __restrict__ w2,
                                                 const float* __restrict__ bias2,
                                                 const int* __restrict__ counts,
                                                 const int* __restrict__ offsets,
                                                 float* __restrict__ pairout) {
  __shared__ _Float16 a_sh[2][TM * LDA];
  __shared__ _Float16 b_sh[2][TN * LDB];
  const int e = blockIdx.x >> 5;
  const int rt = blockIdx.x & 31;
  const int n_e = counts[e];
  const int rbase = rt * TM;
  if (rbase >= n_e) return;
  const int off_e = offsets[e];
  const int nbase = blockIdx.y * TN;
  const int tid = threadIdx.x;
  const int lane = tid & 31;
  const int wave = tid >> 5;
  const int wm = wave & 3;
  const int wn = wave >> 2;

  const int ar = tid >> 1;
  const int ac = (tid & 1) << 4;
  const int arow = (rbase + ar < n_e) ? (rbase + ar) : (n_e - 1);
  const _Float16* asrc = hbuf + (size_t)(off_e + arow) * H_DIM;
  const int bn = tid >> 1;
  const int bk = (tid & 1) << 4;
  const float* bsrc = w2 + ((size_t)e * D_DIM + nbase + bn) * H_DIM;

  const v8f vzero = {0.f, 0.f, 0.f, 0.f, 0.f, 0.f, 0.f, 0.f};
  v8f c[2][4];
#pragma unroll
  for (int mi = 0; mi < 2; ++mi)
#pragma unroll
    for (int ni = 0; ni < 4; ++ni) c[mi][ni] = vzero;

  uint4 pa0 = *(const uint4*)(asrc + ac);
  uint4 pa1 = *(const uint4*)(asrc + ac + 8);
  float4 pb0 = *(const float4*)(bsrc + bk + 0);
  float4 pb1 = *(const float4*)(bsrc + bk + 4);
  float4 pb2 = *(const float4*)(bsrc + bk + 8);
  float4 pb3 = *(const float4*)(bsrc + bk + 12);
  *(uint4*)(a_sh[0] + ar * LDA + ac)     = pa0;
  *(uint4*)(a_sh[0] + ar * LDA + ac + 8) = pa1;
  cvt_store8(b_sh[0] + bn * LDB + bk,     pb0, pb1);
  cvt_store8(b_sh[0] + bn * LDB + bk + 8, pb2, pb3);
  __syncthreads();

  for (int kb = 0; kb < H_DIM; kb += KB) {
    const int p = (kb >> 5) & 1;
    const bool more = (kb + KB) < H_DIM;
    if (more) {
      pa0 = *(const uint4*)(asrc + kb + KB + ac);
      pa1 = *(const uint4*)(asrc + kb + KB + ac + 8);
      pb0 = *(const float4*)(bsrc + kb + KB + bk + 0);
      pb1 = *(const float4*)(bsrc + kb + KB + bk + 4);
      pb2 = *(const float4*)(bsrc + kb + KB + bk + 8);
      pb3 = *(const float4*)(bsrc + kb + KB + bk + 12);
      __builtin_prefetch(bsrc + kb + 2 * KB, 0, 0);
    }
    v16h a0 = load_a_frag(a_sh[p], lane, 32 * wm);
    v16h a1 = load_a_frag(a_sh[p], lane, 32 * wm + 16);
#pragma unroll
    for (int ni = 0; ni < 4; ++ni) {
      v16h bf = load_b_frag(b_sh[p], lane, 64 * wn + 16 * ni);
      c[0][ni] = __builtin_amdgcn_wmma_f32_16x16x32_f16(false, a0, false, bf, (short)0, c[0][ni], false, false);
      c[1][ni] = __builtin_amdgcn_wmma_f32_16x16x32_f16(false, a1, false, bf, (short)0, c[1][ni], false, false);
    }
    if (more) {
      *(uint4*)(a_sh[p ^ 1] + ar * LDA + ac)     = pa0;
      *(uint4*)(a_sh[p ^ 1] + ar * LDA + ac + 8) = pa1;
      cvt_store8(b_sh[p ^ 1] + bn * LDB + bk,     pb0, pb1);
      cvt_store8(b_sh[p ^ 1] + bn * LDB + bk + 8, pb2, pb3);
      __syncthreads();
    }
  }

  const int nl = lane & 15;
  const int gh = (lane < 16) ? 0 : 8;
#pragma unroll
  for (int mi = 0; mi < 2; ++mi)
#pragma unroll
    for (int ni = 0; ni < 4; ++ni) {
      int col = nbase + 64 * wn + 16 * ni + nl;
      float bias = bias2[e * D_DIM + col];
#pragma unroll
      for (int g = 0; g < 8; ++g) {
        int r = rbase + 32 * wm + 16 * mi + gh + g;
        if (r < n_e) {
          pairout[(size_t)(off_e + r) * D_DIM + col] = c[mi][ni][g] + bias;
        }
      }
    }
}

// ---- combine: out[t] = w0*pairout[r0] + w1*pairout[r1] (deterministic, no atomics) ----
__global__ __launch_bounds__(256) void moe_combine(const int* __restrict__ tok_slot,
                                                   const float* __restrict__ tok_w,
                                                   const int* __restrict__ offsets,
                                                   const float* __restrict__ pairout,
                                                   float* __restrict__ out) {
  int t = blockIdx.x;
  int s0 = tok_slot[2 * t], s1 = tok_slot[2 * t + 1];
  float w0 = tok_w[2 * t], w1 = tok_w[2 * t + 1];
  int r0 = offsets[s0 >> 12] + (s0 & (T_TOK - 1));
  int r1 = offsets[s1 >> 12] + (s1 & (T_TOK - 1));
  int d = threadIdx.x * 4;
  float4 A = *(const float4*)(pairout + (size_t)r0 * D_DIM + d);
  float4 B = *(const float4*)(pairout + (size_t)r1 * D_DIM + d);
  float4 o;
  o.x = w0 * A.x + w1 * B.x;
  o.y = w0 * A.y + w1 * B.y;
  o.z = w0 * A.z + w1 * B.z;
  o.w = w0 * A.w + w1 * B.w;
  *(float4*)(out + (size_t)t * D_DIM + d) = o;
}

extern "C" void kernel_launch(void* const* d_in, const int* in_sizes, int n_in,
                              void* d_out, int out_size, void* d_ws, size_t ws_size,
                              hipStream_t stream) {
  const float* x  = (const float*)d_in[0];
  const float* rw = (const float*)d_in[1];
  const float* w1 = (const float*)d_in[2];
  const float* b1 = (const float*)d_in[3];
  const float* w2 = (const float*)d_in[4];
  const float* b2 = (const float*)d_in[5];
  float* out = (float*)d_out;

  char* ws = (char*)d_ws;
  int* counts  = (int*)(ws + 0);                       // E
  int* offsets = (int*)(ws + 64);                      // E+1
  int* pair_tok = (int*)(ws + 128);                    // E*T
  size_t o = 128 + (size_t)E_NUM * T_TOK * 4;
  int* tok_slot = (int*)(ws + o);  o += (size_t)T_TOK * 2 * 4;   // T*2
  float* tok_w  = (float*)(ws + o); o += (size_t)T_TOK * 2 * 4;  // T*2
  o = (o + 255) & ~(size_t)255;
  _Float16* xh = (_Float16*)(ws + o);   o += (size_t)T_TOK * D_DIM * 2;      // 8 MB
  _Float16* hbuf = (_Float16*)(ws + o); o += (size_t)2 * T_TOK * H_DIM * 2;  // 64 MB
  float* pairout = (float*)(ws + o);                                          // 32 MB

  moe_zero_counts<<<1, 32, 0, stream>>>(counts);
  moe_cvt_x<<<(T_TOK * D_DIM) / 1024, 256, 0, stream>>>(x, xh);
  moe_router<<<T_TOK / 8, 256, 0, stream>>>(x, rw, counts, pair_tok, tok_slot, tok_w);
  moe_scan<<<1, 1, 0, stream>>>(counts, offsets);
  moe_gemm1<<<dim3(E_NUM * (T_TOK / TM), H_DIM / TN), 256, 0, stream>>>(
      xh, w1, b1, pair_tok, counts, offsets, hbuf);
  moe_gemm2<<<dim3(E_NUM * (T_TOK / TM), D_DIM / TN), 256, 0, stream>>>(
      hbuf, w2, b2, counts, offsets, pairout);
  moe_combine<<<T_TOK, 256, 0, stream>>>(tok_slot, tok_w, offsets, pairout, out);
}
